// NegEuclidDist_52888227283652
// MI455X (gfx1250) — matrix operand order repaired and checked by reading, not compiled, not run on hardware
//
#include <hip/hip_runtime.h>

// ---------------------------------------------------------------------------
// NegEuclidDist: out[n,c] = 2*(X @ P^T) - ||x_n||^2 - ||p_c||^2
// X: [N,D] f32, P: [C,D] f32, out: [N,C] f32.  N=8192, D=1024, C=4096.
// f16 WMMA GEMM (f32 accumulate), double-buffered LDS, fused epilogue.
// Block tile 128x256, 8 waves, wave tile 64x64 (16 WMMAs / K-step / wave).
// ---------------------------------------------------------------------------

static constexpr int Nn = 8192;
static constexpr int Dd = 1024;
static constexpr int Cc = 4096;

static constexpr int BM  = 128;  // block tile M (X rows)
static constexpr int BN  = 256;  // block tile N (P rows)
static constexpr int BK  = 32;   // K step (one f16 WMMA K)
static constexpr int TPB = 256;  // 8 waves (wave32)
static constexpr int LDH = 40;   // LDS row stride in halves (64B data + 16B pad)

typedef _Float16 v8h  __attribute__((ext_vector_type(8)));
typedef _Float16 v16h __attribute__((ext_vector_type(16)));
typedef float    v8f  __attribute__((ext_vector_type(8)));

// ---------------------------------------------------------------------------
// Row-norm kernel: ws[r] = sum(src[r,:]^2); rows [0,N) from X, [N,N+C) from P
// ---------------------------------------------------------------------------
__global__ __launch_bounds__(256) void negdist_norms(const float* __restrict__ x,
                                                     const float* __restrict__ p,
                                                     float* __restrict__ ws) {
    const int row = blockIdx.x;
    const float* src = (row < Nn) ? (x + (size_t)row * Dd)
                                  : (p + (size_t)(row - Nn) * Dd);
    float s = 0.0f;
    const float4* src4 = (const float4*)src;
    for (int i = threadIdx.x; i < Dd / 4; i += blockDim.x) {
        float4 v = src4[i];
        s += v.x * v.x + v.y * v.y + v.z * v.z + v.w * v.w;
    }
    for (int off = 16; off > 0; off >>= 1) s += __shfl_xor(s, off, 32);
    __shared__ float red[8];
    if ((threadIdx.x & 31) == 0) red[threadIdx.x >> 5] = s;
    __syncthreads();
    if (threadIdx.x == 0) {
        float t = 0.0f;
        for (int w = 0; w < 8; ++w) t += red[w];
        ws[row] = t;
    }
}

// ---------------------------------------------------------------------------
// Fragment loaders (layouts per CDNA5 ISA 7.12.2)
// A 16x32 f16: lane L -> row L&15; lanes 0-15 hold K 0-7 & 16-23,
//              lanes 16-31 hold K 8-15 & 24-31 (two separated 16B chunks).
// B 32x16 f16: lane L -> col L&15; lanes 0-15 hold K 0-7 & 16-23,
//              lanes 16-31 hold K 8-15 & 24-31 (same order as A).
// ---------------------------------------------------------------------------
__device__ __forceinline__ v16h ldsA_frag(const _Float16* __restrict__ fr, int lane) {
    const int r  = lane & 15;
    const int hi = lane >> 4;
    const _Float16* q = fr + r * LDH + hi * 8;
    v8h lo = *(const v8h*)(q);        // K: hi*8 .. hi*8+7
    v8h up = *(const v8h*)(q + 16);   // K: 16+hi*8 .. 16+hi*8+7
    return __builtin_shufflevector(lo, up, 0,1,2,3,4,5,6,7,8,9,10,11,12,13,14,15);
}

__device__ __forceinline__ v16h ldsB_frag(const _Float16* __restrict__ fr, int lane) {
    const int c  = lane & 15;
    const int hi = lane >> 4;
    const _Float16* q = fr + c * LDH + hi * 8;
    v8h lo = *(const v8h*)(q);        // K: hi*8 .. hi*8+7
    v8h up = *(const v8h*)(q + 16);   // K: 16+hi*8 .. 16+hi*8+7
    return __builtin_shufflevector(lo, up, 0,1,2,3,4,5,6,7,8,9,10,11,12,13,14,15);
}

__device__ __forceinline__ v8h cvt8(const float4& a, const float4& b) {
    return v8h{ (_Float16)a.x, (_Float16)a.y, (_Float16)a.z, (_Float16)a.w,
                (_Float16)b.x, (_Float16)b.y, (_Float16)b.z, (_Float16)b.w };
}

// ---------------------------------------------------------------------------
// GEMM + fused epilogue
// ---------------------------------------------------------------------------
__global__ __launch_bounds__(TPB) void negdist_gemm(const float* __restrict__ X,
                                                    const float* __restrict__ P,
                                                    const float* __restrict__ xsq,
                                                    const float* __restrict__ psq,
                                                    float* __restrict__ out) {
    __shared__ __align__(16) _Float16 smA[2][BM * LDH];  // 2 x 10 KB
    __shared__ __align__(16) _Float16 smB[2][BN * LDH];  // 2 x 20 KB

    const int t    = threadIdx.x;
    const int lane = t & 31;
    const int wid  = t >> 5;
    const int wm   = wid >> 2;   // 0..1 -> 64-row slab of BM
    const int wn   = wid & 3;    // 0..3 -> 64-col slab of BN

    const size_t rowA0 = (size_t)blockIdx.y * BM;
    const size_t rowB0 = (size_t)blockIdx.x * BN;

    // Staging registers: 8-float chunks (row-contiguous, 32B global / 16B LDS)
    float4 rA[2][2], rB[4][2];

    auto stage = [&](int kt) {
        const size_t kbase = (size_t)kt * BK;
#pragma unroll
        for (int it = 0; it < 2; ++it) {          // A: 512 chunks / 256 thr
            const int ch = t + it * TPB;
            const int r  = ch >> 2;               // 0..127
            const int c8 = ch & 3;                // 8-float chunk in row
            const float* g = X + (rowA0 + r) * Dd + kbase + c8 * 8;
            rA[it][0] = *(const float4*)(g);
            rA[it][1] = *(const float4*)(g + 4);
        }
#pragma unroll
        for (int it = 0; it < 4; ++it) {          // B: 1024 chunks / 256 thr
            const int ch = t + it * TPB;
            const int r  = ch >> 2;               // 0..255
            const int c8 = ch & 3;
            const float* g = P + (rowB0 + r) * Dd + kbase + c8 * 8;
            rB[it][0] = *(const float4*)(g);
            rB[it][1] = *(const float4*)(g + 4);
        }
    };
    auto commit = [&](int buf) {
#pragma unroll
        for (int it = 0; it < 2; ++it) {
            const int ch = t + it * TPB;
            const int r  = ch >> 2;
            const int c8 = ch & 3;
            *(v8h*)(&smA[buf][r * LDH + c8 * 8]) = cvt8(rA[it][0], rA[it][1]);
        }
#pragma unroll
        for (int it = 0; it < 4; ++it) {
            const int ch = t + it * TPB;
            const int r  = ch >> 2;
            const int c8 = ch & 3;
            *(v8h*)(&smB[buf][r * LDH + c8 * 8]) = cvt8(rB[it][0], rB[it][1]);
        }
    };

    stage(0);
    commit(0);
    __syncthreads();

    v8f acc[4][4] = {};
    constexpr int KT = Dd / BK;   // 32 K-steps

    for (int kt = 0; kt < KT; ++kt) {
        const int cur = kt & 1;
        if (kt + 1 < KT) stage(kt + 1);   // global loads overlap WMMAs below

        const _Float16* sA = &smA[cur][(wm * 64) * LDH];
        const _Float16* sB = &smB[cur][(wn * 64) * LDH];

        v16h a[4], b[4];
#pragma unroll
        for (int i = 0; i < 4; ++i) a[i] = ldsA_frag(sA + i * 16 * LDH, lane);
#pragma unroll
        for (int j = 0; j < 4; ++j) b[j] = ldsB_frag(sB + j * 16 * LDH, lane);

#pragma unroll
        for (int i = 0; i < 4; ++i)
#pragma unroll
            for (int j = 0; j < 4; ++j)
                acc[i][j] = __builtin_amdgcn_wmma_f32_16x16x32_f16(
                    /*neg_a=*/false, a[i], /*neg_b=*/false, b[j],
                    /*c_mod=*/(short)0, acc[i][j],
                    /*reuse_a=*/false, /*reuse_b=*/false);

        if (kt + 1 < KT) commit(cur ^ 1);  // fill the other buffer
        __syncthreads();
    }

    // Epilogue: C/D layout -> lane holds col = lane&15, rows (lane>>4)*8 + v.
    // Non-temporal stores: out is write-once, keep X/P resident in L2.
    const int col16 = lane & 15;
    const int rhalf = (lane >> 4) * 8;
#pragma unroll
    for (int i = 0; i < 4; ++i) {
#pragma unroll
        for (int j = 0; j < 4; ++j) {
            const size_t gn  = rowB0 + wn * 64 + j * 16 + col16;
            const size_t gm0 = rowA0 + wm * 64 + i * 16 + rhalf;
            const float pn = psq[gn];
#pragma unroll
            for (int v = 0; v < 8; ++v) {
                const size_t gm = gm0 + v;
                const float val = 2.0f * acc[i][j][v] - xsq[gm] - pn;
                __builtin_nontemporal_store(val, &out[gm * Cc + gn]);
            }
        }
    }
}

// ---------------------------------------------------------------------------
extern "C" void kernel_launch(void* const* d_in, const int* in_sizes, int n_in,
                              void* d_out, int out_size, void* d_ws, size_t ws_size,
                              hipStream_t stream) {
    const float* x = (const float*)d_in[0];   // [N, D]
    const float* p = (const float*)d_in[1];   // [C, D]
    float* out = (float*)d_out;               // [N, C]
    float* ws  = (float*)d_ws;                // [N + C] row norms

    negdist_norms<<<Nn + Cc, 256, 0, stream>>>(x, p, ws);

    dim3 grid(Cc / BN, Nn / BM);   // 16 x 64 blocks
    negdist_gemm<<<grid, TPB, 0, stream>>>(x, p, ws, ws + Nn, out);
}
